// SeparateKV_63402307224429
// MI455X (gfx1250) — matrix operand order repaired and checked by reading, not compiled, not run on hardware
//
#include <hip/hip_runtime.h>
#include <hip/hip_bf16.h>

// ---------------------------------------------------------------------------
// Types
// ---------------------------------------------------------------------------
typedef __bf16 bf16_t;
typedef __attribute__((ext_vector_type(16))) __bf16 v16bf;
typedef __attribute__((ext_vector_type(8)))  __bf16 v8bf;
typedef __attribute__((ext_vector_type(8)))  float  v8f;
typedef __attribute__((ext_vector_type(4)))  unsigned int u32x4;

// Problem constants (match reference)
#define BB   2
#define SS   2048
#define HID  2048
#define HH   32
#define HD   64
#define MQA  64
#define QR   2048
#define KR   448
#define VR   512

// ---------------------------------------------------------------------------
// Helpers
// ---------------------------------------------------------------------------
__device__ inline v16bf ld16x2(const bf16_t* p0, const bf16_t* p1) {
  v8bf lo = *(const v8bf*)p0;
  v8bf hi = *(const v8bf*)p1;
  return __builtin_shufflevector(lo, hi, 0,1,2,3,4,5,6,7,8,9,10,11,12,13,14,15);
}

// CDNA5 async copy: global -> LDS, 16 bytes per lane, tracked by ASYNCcnt.
// LDS destination address = low 32 bits of the generic shared pointer
// (flat LDS addressing truncates to the wave-relative LDS offset).
__device__ inline void async_cp_b128(void* lds_dst, const void* gsrc) {
  unsigned int loff = (unsigned int)(uintptr_t)lds_dst;
  asm volatile("global_load_async_to_lds_b128 %0, %1, off"
               :: "v"(loff), "v"(gsrc) : "memory");
}
#define WAIT_ASYNC(n) asm volatile("s_wait_asynccnt %0" :: "n"(n) : "memory")

// ---------------------------------------------------------------------------
// f32 -> bf16 conversion
// ---------------------------------------------------------------------------
__global__ void mla_cvt_bf16(const float* __restrict__ in, bf16_t* __restrict__ out, size_t n) {
  size_t i = (size_t)blockIdx.x * blockDim.x + threadIdx.x;
  if (i < n) out[i] = (bf16_t)in[i];
}

// ---------------------------------------------------------------------------
// Generic NT GEMM:  C[M,N] = A[M,K] * B[N,K]^T   (bf16 in, f32 acc)
// block tile 128x64 (MxN), 128 threads (4 waves), K-step 32, WMMA bf16.
// Each wave computes a 32x64 sub-tile: 2 A-frags x 4 B-frags = 8 WMMA/K-step.
// Async (ASYNCcnt) double-buffered LDS staging.
// ---------------------------------------------------------------------------
__global__ __launch_bounds__(128) void mla_gemm_nt(
    const bf16_t* __restrict__ A, const bf16_t* __restrict__ Bm,
    void* __restrict__ Cout, int M, int N, int K,
    int lda, int ldb, int ldc, int out_f32)
{
  __shared__ bf16_t As[2][128][40];  // 32 + 8 pad (80B rows, 16B aligned)
  __shared__ bf16_t Bs[2][64][40];

  const int tid  = threadIdx.x;
  const int wave = tid >> 5;
  const int lane = tid & 31;
  const int l16  = lane & 15;
  const int hi   = lane >> 4;
  const int m0   = blockIdx.y * 128;
  const int n0   = blockIdx.x * 64;

  // Staging: A tile 128x32 = 512 chunks of 8 bf16 (4/thread),
  //          B tile  64x32 = 256 chunks           (2/thread).
  int ar[4], ac[4], br[2], bc[2];
#pragma unroll
  for (int i = 0; i < 4; ++i) {
    int cc = tid + i * 128;            // [0,512)
    ar[i] = cc >> 2; ac[i] = (cc & 3) * 8;
  }
#pragma unroll
  for (int i = 0; i < 2; ++i) {
    int cc = tid + i * 128;            // [0,256)
    br[i] = cc >> 2; bc[i] = (cc & 3) * 8;
  }

  const int nk = K >> 5;
  v8f acc[2][4] = {};

  // prologue: stage tile 0 into buffer 0
#pragma unroll
  for (int i = 0; i < 4; ++i)
    async_cp_b128(&As[0][ar[i]][ac[i]], &A[(size_t)(m0 + ar[i]) * lda + ac[i]]);
#pragma unroll
  for (int i = 0; i < 2; ++i)
    async_cp_b128(&Bs[0][br[i]][bc[i]], &Bm[(size_t)(n0 + br[i]) * ldb + bc[i]]);

  for (int kk = 0; kk < nk; ++kk) {
    const int buf = kk & 1;
    if (kk + 1 < nk) {
      // stage next tile into the other buffer, then wait for current tile
      const int k1 = (kk + 1) << 5;
#pragma unroll
      for (int i = 0; i < 4; ++i)
        async_cp_b128(&As[buf ^ 1][ar[i]][ac[i]],
                      &A[(size_t)(m0 + ar[i]) * lda + k1 + ac[i]]);
#pragma unroll
      for (int i = 0; i < 2; ++i)
        async_cp_b128(&Bs[buf ^ 1][br[i]][bc[i]],
                      &Bm[(size_t)(n0 + br[i]) * ldb + k1 + bc[i]]);
      WAIT_ASYNC(6);           // in-order completion: current tile is in LDS
    } else {
      WAIT_ASYNC(0);
    }
    __syncthreads();

    // A fragments: wave rows [32*wave, 32*wave+32), K-chunk of 32
    v16bf afrag[2];
#pragma unroll
    for (int mi = 0; mi < 2; ++mi) {
      const int arow = wave * 32 + mi * 16 + l16;
      afrag[mi] = ld16x2(&As[buf][arow][hi * 8], &As[buf][arow][16 + hi * 8]);
    }

#pragma unroll
    for (int t = 0; t < 4; ++t) {
      const int brow = t * 16 + l16;
      v16bf bfrag = ld16x2(&Bs[buf][brow][hi * 8], &Bs[buf][brow][16 + hi * 8]);
#pragma unroll
      for (int mi = 0; mi < 2; ++mi)
        acc[mi][t] = __builtin_amdgcn_wmma_f32_16x16x32_bf16(
            false, afrag[mi], false, bfrag, (short)0, acc[mi][t], false, false);
    }
    __syncthreads();           // all waves done with buf before it is refilled
  }

#pragma unroll
  for (int mi = 0; mi < 2; ++mi)
#pragma unroll
    for (int t = 0; t < 4; ++t)
#pragma unroll
      for (int r = 0; r < 8; ++r) {
        int row = m0 + wave * 32 + mi * 16 + hi * 8 + r;
        int col = n0 + t * 16 + l16;
        float v = acc[mi][t][r];
        if (out_f32) ((float*)Cout)[(size_t)row * ldc + col] = v;
        else         ((bf16_t*)Cout)[(size_t)row * ldc + col] = (bf16_t)v;
      }
}

// ---------------------------------------------------------------------------
// RoPE on q (in place, bf16 buffer laid out (B,S,H,128); rope dims are [0,64))
// one thread handles the pair (d, d+32) -> race free
// ---------------------------------------------------------------------------
__global__ void mla_rope_q(bf16_t* __restrict__ q,
                           const float* __restrict__ cosb,
                           const float* __restrict__ sinb)
{
  size_t idx = (size_t)blockIdx.x * blockDim.x + threadIdx.x;
  size_t total = (size_t)BB * SS * HH * 32;
  if (idx >= total) return;
  int d = idx & 31; size_t t = idx >> 5;
  int h = t % HH; t /= HH;
  int s = t % SS; int b = (int)(t / SS);
  size_t base = (((size_t)b * SS + s) * HH + h) * 128;
  size_t cs   = ((size_t)b * SS + s) * MQA;
  float x1 = (float)q[base + d];
  float x2 = (float)q[base + d + 32];
  float c1 = cosb[cs + d],      s1 = sinb[cs + d];
  float c2 = cosb[cs + d + 32], s2 = sinb[cs + d + 32];
  q[base + d]      = (bf16_t)(x1 * c1 - x2 * s1);
  q[base + d + 32] = (bf16_t)(x2 * c2 + x1 * s2);
}

// ---------------------------------------------------------------------------
// Build K (B,H,S,128): rope(k_a[:, :64]) broadcast across heads + k_nope
// ---------------------------------------------------------------------------
__global__ void mla_build_k(const bf16_t* __restrict__ ka,     // (B*S, 512)
                            const bf16_t* __restrict__ knope,  // (B*S, H*64)
                            const float* __restrict__ cosb,
                            const float* __restrict__ sinb,
                            bf16_t* __restrict__ Kb)           // (B,H,S,128)
{
  size_t idx = (size_t)blockIdx.x * blockDim.x + threadIdx.x;
  size_t total = (size_t)BB * HH * SS * 96;
  if (idx >= total) return;
  int j = idx % 96; size_t t = idx / 96;
  int s = t % SS; t /= SS;
  int h = t % HH; int b = (int)(t / HH);
  size_t krow = ((size_t)b * SS + s);
  size_t out  = (((size_t)(b * HH + h) * SS) + s) * 128;
  if (j < 32) {
    size_t cs = krow * MQA;
    float x1 = (float)ka[krow * 512 + j];
    float x2 = (float)ka[krow * 512 + j + 32];
    float c1 = cosb[cs + j],      s1 = sinb[cs + j];
    float c2 = cosb[cs + j + 32], s2 = sinb[cs + j + 32];
    Kb[out + j]      = (bf16_t)(x1 * c1 - x2 * s1);
    Kb[out + j + 32] = (bf16_t)(x2 * c2 + x1 * s2);
  } else {
    int d = j + 32;                       // 64..127
    Kb[out + d] = knope[krow * (HH * HD) + h * HD + (d - 64)];
  }
}

// ---------------------------------------------------------------------------
// Build Vt (B,H,64,S) from v (B,S,H*64)
// ---------------------------------------------------------------------------
__global__ void mla_build_vt(const bf16_t* __restrict__ v, bf16_t* __restrict__ Vt)
{
  size_t idx = (size_t)blockIdx.x * blockDim.x + threadIdx.x;
  size_t total = (size_t)BB * HH * HD * SS;
  if (idx >= total) return;
  int s = idx % SS; size_t t = idx / SS;
  int d = t % HD; t /= HD;
  int h = t % HH; int b = (int)(t / HH);
  Vt[idx] = v[(((size_t)b * SS + s) * HH + h) * HD + d];
}

// ---------------------------------------------------------------------------
// Flash attention, causal.  One block = one (b,h) x 64 query rows.
// 128 threads / 4 waves; wave w owns query rows [16w,16w+16).
// scores and P*V both via v_wmma_f32_16x16x32_bf16; online softmax via LDS.
// Q/K/V tiles staged with async-to-LDS copies.
// ---------------------------------------------------------------------------
__global__ __launch_bounds__(128) void mla_flash(
    const bf16_t* __restrict__ Qb,   // (B,S,H,128)
    const bf16_t* __restrict__ Kb,   // (B,H,S,128)
    const bf16_t* __restrict__ Vt,   // (B,H,64,S)
    bf16_t* __restrict__ Ob)         // (B,S,H,64)
{
  __shared__ bf16_t Qs[64][136];
  __shared__ bf16_t Ks[64][136];
  __shared__ bf16_t Vs[64][72];
  __shared__ float  Ss[64][72];
  __shared__ bf16_t Ps[64][72];
  __shared__ float  mS[64], lS[64], aS[64];

  const int tid  = threadIdx.x;
  const int wave = tid >> 5;
  const int lane = tid & 31;
  const int l16  = lane & 15;
  const int hi   = lane >> 4;
  const int bh   = blockIdx.x;           // b*H + h
  const int b    = bh / HH;
  const int h    = bh % HH;
  const int q0   = blockIdx.y * 64;

  // ---- stage Q tile (64 x 128) from (B,S,H,128) layout
#pragma unroll
  for (int i = 0; i < 8; ++i) {
    int cc  = tid + i * 128;             // [0,1024)
    int row = cc >> 4;
    int c8  = (cc & 15) * 8;
    async_cp_b128(&Qs[row][c8],
                  &Qb[(((size_t)b * SS + q0 + row) * HH + h) * 128 + c8]);
  }
  if (tid < 64) { mS[tid] = -3.0e38f; lS[tid] = 0.0f; }
  WAIT_ASYNC(0);
  __syncthreads();

  // preload Q fragments (4 K-chunks of 32)
  v16bf aq[4];
#pragma unroll
  for (int kc = 0; kc < 4; ++kc) {
    const bf16_t* qr = &Qs[wave * 16 + l16][0];
    aq[kc] = ld16x2(qr + kc * 32 + hi * 8, qr + kc * 32 + 16 + hi * 8);
  }

  v8f o[4] = {};

  for (int j0 = 0; j0 <= q0; j0 += 64) {
    // ---- stage K tile (64 x 128) and V^T tile (64d x 64kv)
#pragma unroll
    for (int i = 0; i < 8; ++i) {
      int cc  = tid + i * 128;
      int row = cc >> 4;
      int c8  = (cc & 15) * 8;
      async_cp_b128(&Ks[row][c8],
                    &Kb[((size_t)bh * SS + j0 + row) * 128 + c8]);
    }
#pragma unroll
    for (int i = 0; i < 4; ++i) {
      int cc  = tid + i * 128;           // [0,512)
      int row = cc >> 3;
      int c8  = (cc & 7) * 8;
      async_cp_b128(&Vs[row][c8],
                    &Vt[((size_t)bh * 64 + row) * SS + j0 + c8]);
    }
    WAIT_ASYNC(0);
    __syncthreads();

    // ---- S = Q K^T  (64x64, f32)
    v8f s[4] = {};
#pragma unroll
    for (int kc = 0; kc < 4; ++kc) {
#pragma unroll
      for (int t = 0; t < 4; ++t) {
        const bf16_t* kr = &Ks[t * 16 + l16][0];
        v16bf bk = ld16x2(kr + kc * 32 + hi * 8, kr + kc * 32 + 16 + hi * 8);
        s[t] = __builtin_amdgcn_wmma_f32_16x16x32_bf16(
            false, aq[kc], false, bk, (short)0, s[t], false, false);
      }
    }
#pragma unroll
    for (int t = 0; t < 4; ++t)
#pragma unroll
      for (int r = 0; r < 8; ++r)
        Ss[wave * 16 + hi * 8 + r][t * 16 + l16] = s[t][r] * 0.125f; // 1/sqrt(64)
    __syncthreads();

    // ---- online softmax (one thread per query row)
    if (tid < 64) {
      const int i  = tid;
      const int gi = q0 + i;
      float mold = mS[i];
      float mnew = mold;
      for (int jj = 0; jj < 64; ++jj) {
        float x = (j0 + jj > gi) ? -1.0e9f : Ss[i][jj];
        mnew = fmaxf(mnew, x);
      }
      float alpha = __expf(mold - mnew);
      float lsum  = 0.0f;
      for (int jj = 0; jj < 64; ++jj) {
        float x = (j0 + jj > gi) ? -1.0e9f : Ss[i][jj];
        float p = __expf(x - mnew);
        lsum += p;
        Ps[i][jj] = (bf16_t)p;
      }
      lS[i] = lS[i] * alpha + lsum;
      mS[i] = mnew;
      aS[i] = alpha;
    }
    __syncthreads();

    // ---- rescale O accumulators by alpha(row)
    float al[8];
#pragma unroll
    for (int r = 0; r < 8; ++r) al[r] = aS[wave * 16 + hi * 8 + r];
#pragma unroll
    for (int t = 0; t < 4; ++t)
#pragma unroll
      for (int r = 0; r < 8; ++r) o[t][r] *= al[r];

    // ---- O += P V   (A = P 64x64, B = V^T as (N=d64, K=kv64))
#pragma unroll
    for (int kc = 0; kc < 2; ++kc) {
      const bf16_t* pr = &Ps[wave * 16 + l16][0];
      v16bf ap = ld16x2(pr + kc * 32 + hi * 8, pr + kc * 32 + 16 + hi * 8);
#pragma unroll
      for (int t = 0; t < 4; ++t) {
        const bf16_t* vr = &Vs[t * 16 + l16][0];
        v16bf bv = ld16x2(vr + kc * 32 + hi * 8, vr + kc * 32 + 16 + hi * 8);
        o[t] = __builtin_amdgcn_wmma_f32_16x16x32_bf16(
            false, ap, false, bv, (short)0, o[t], false, false);
      }
    }
    __syncthreads();
  }

  // ---- epilogue: divide by l, write (B,S,H,64) bf16
  float linv[8];
#pragma unroll
  for (int r = 0; r < 8; ++r) linv[r] = 1.0f / lS[wave * 16 + hi * 8 + r];
#pragma unroll
  for (int t = 0; t < 4; ++t)
#pragma unroll
    for (int r = 0; r < 8; ++r) {
      int row = q0 + wave * 16 + hi * 8 + r;
      int d   = t * 16 + l16;
      Ob[(((size_t)b * SS + row) * HH + h) * HD + d] = (bf16_t)(o[t][r] * linv[r]);
    }
}

// ---------------------------------------------------------------------------
// Host launcher
// ---------------------------------------------------------------------------
extern "C" void kernel_launch(void* const* d_in, const int* in_sizes, int n_in,
                              void* d_out, int out_size, void* d_ws, size_t ws_size,
                              hipStream_t stream) {
  const float* hs   = (const float*)d_in[0];
  /* d_in[1] attention_mask: causal, applied analytically */
  const float* cosb = (const float*)d_in[2];
  const float* sinb = (const float*)d_in[3];
  const float* qaW  = (const float*)d_in[4];
  const float* qbW  = (const float*)d_in[5];
  const float* kaW  = (const float*)d_in[6];
  const float* kbW  = (const float*)d_in[7];
  const float* vaW  = (const float*)d_in[8];
  const float* vbW  = (const float*)d_in[9];
  const float* oW   = (const float*)d_in[10];

  char* w = (char*)d_ws;
  auto carve = [&](size_t elems) -> bf16_t* {
    bf16_t* p = (bf16_t*)w;
    w += ((elems * sizeof(bf16_t) + 255) / 256) * 256;
    return p;
  };

  const size_t BS = (size_t)BB * SS;                 // 4096
  bf16_t* hsb   = carve(BS * HID);                   // hidden bf16
  bf16_t* Wqa   = carve((size_t)QR * HID);
  bf16_t* Wqb   = carve((size_t)HH * (MQA + HD) * QR);
  bf16_t* Wka   = carve((size_t)(MQA + KR) * HID);
  bf16_t* Wkb   = carve((size_t)HH * HD * KR);
  bf16_t* Wva   = carve((size_t)VR * HID);
  bf16_t* Wvb   = carve((size_t)HH * HD * VR);
  bf16_t* Wo    = carve((size_t)HID * HH * HD);
  bf16_t* qlat  = carve(BS * QR);
  bf16_t* qfull = carve(BS * HH * 128);              // (B,S,H,128)
  bf16_t* ka    = carve(BS * (MQA + KR));            // (B*S, 512)
  bf16_t* knope = carve(BS * HH * HD);
  bf16_t* Kbuf  = carve((size_t)BB * HH * SS * 128); // (B,H,S,128)
  bf16_t* vlat  = carve(BS * VR);
  bf16_t* vfull = carve(BS * HH * HD);
  bf16_t* Vt    = carve((size_t)BB * HH * HD * SS);  // (B,H,64,S)
  bf16_t* attnb = carve(BS * HH * HD);               // (B,S,H*64)

  auto cvt = [&](const float* src, bf16_t* dst, size_t n) {
    mla_cvt_bf16<<<(unsigned)((n + 255) / 256), 256, 0, stream>>>(src, dst, n);
  };
  cvt(hs,  hsb, BS * HID);
  cvt(qaW, Wqa, (size_t)QR * HID);
  cvt(qbW, Wqb, (size_t)HH * (MQA + HD) * QR);
  cvt(kaW, Wka, (size_t)(MQA + KR) * HID);
  cvt(kbW, Wkb, (size_t)HH * HD * KR);
  cvt(vaW, Wva, (size_t)VR * HID);
  cvt(vbW, Wvb, (size_t)HH * HD * VR);
  cvt(oW,  Wo,  (size_t)HID * HH * HD);

  auto gemm = [&](const bf16_t* A, const bf16_t* Bm, void* C,
                  int M, int N, int K, int lda, int ldb, int ldc, int outf32) {
    dim3 g(N / 64, M / 128);
    mla_gemm_nt<<<g, 128, 0, stream>>>(A, Bm, C, M, N, K, lda, ldb, ldc, outf32);
  };

  // q = hs @ q_a^T @ q_b^T
  gemm(hsb, Wqa, qlat, (int)BS, QR, HID, HID, HID, QR, 0);
  gemm(qlat, Wqb, qfull, (int)BS, HH * 128, QR, QR, QR, HH * 128, 0);
  // k_a = hs @ k_a^T ; k_nope = k_a[:,64:] @ k_b^T
  gemm(hsb, Wka, ka, (int)BS, MQA + KR, HID, HID, HID, MQA + KR, 0);
  gemm(ka + MQA, Wkb, knope, (int)BS, HH * HD, KR, MQA + KR, KR, HH * HD, 0);
  // v = hs @ v_a^T @ v_b^T
  gemm(hsb, Wva, vlat, (int)BS, VR, HID, HID, HID, VR, 0);
  gemm(vlat, Wvb, vfull, (int)BS, HH * HD, VR, VR, VR, HH * HD, 0);

  // RoPE + layout builds
  {
    size_t n = (size_t)BB * SS * HH * 32;
    mla_rope_q<<<(unsigned)((n + 255) / 256), 256, 0, stream>>>(qfull, cosb, sinb);
  }
  {
    size_t n = (size_t)BB * HH * SS * 96;
    mla_build_k<<<(unsigned)((n + 255) / 256), 256, 0, stream>>>(ka, knope, cosb, sinb, Kbuf);
  }
  {
    size_t n = (size_t)BB * HH * HD * SS;
    mla_build_vt<<<(unsigned)((n + 255) / 256), 256, 0, stream>>>(vfull, Vt);
  }

  // flash attention
  {
    dim3 g(BB * HH, SS / 64);
    mla_flash<<<g, 128, 0, stream>>>(qfull, Kbuf, Vt, attnb);
  }

  // out = attn @ o_W^T  (f32 result straight to d_out)
  gemm(attnb, Wo, d_out, (int)BS, HID, HH * HD, HH * HD, HID, HID, 1);
}
